// CombinedGCNMLPModel_79929341378819
// MI455X (gfx1250) — matrix operand order repaired and checked by reading, not compiled, not run on hardware
//
#include <hip/hip_runtime.h>
#include <hip/hip_bf16.h>
#include <float.h>

typedef __attribute__((ext_vector_type(16))) __bf16 bf16x16;
typedef __attribute__((ext_vector_type(8)))  __bf16 bf16x8;
typedef __attribute__((ext_vector_type(2)))  __bf16 bf16x2;
typedef __attribute__((ext_vector_type(8)))  float  f32x8;

#define N_NODES  20000
#define N_EDGES  320000
#define N_GRAPHS 512
#define IN_FEATS 128
#define HID      256
#define RDKIT_IN 2048
#define RDK_H1   1024
#define RDK_H2   512
#define COMB_IN  1024   // 2*HID + RDK_H2
#define COMB_H1  1024
#define COMB_H2  512

// ---------------------------------------------------------------- cast f32->bf16
__global__ void cast_f32_bf16(const float* __restrict__ in, __bf16* __restrict__ out, int n) {
    for (int i = blockIdx.x * blockDim.x + threadIdx.x; i < n; i += gridDim.x * blockDim.x)
        out[i] = (__bf16)in[i];
}

// ---------------------------------------------------------------- WMMA GEMM
// C[M,N] = act(A[M,K](bf16) @ B[K,N](bf16) + bias), row-major.
// Block tile 64x128, 256 threads = 8 waves laid out 2(M) x 4(N);
// each wave computes a 32x32 tile = 4 x (16x16) WMMA accumulators, K stepped by 32.
// B tile staged transposed in LDS (packed b32 stores); fragment reads are b128s.
// A rows are clamped (not masked) so the hot loop has no EXEC divergence.
__launch_bounds__(256)
__global__ void gemm_bf16_wmma(const __bf16* __restrict__ A, const __bf16* __restrict__ B,
                               const float* __restrict__ bias,
                               float* __restrict__ C, __bf16* __restrict__ Cb,
                               int M, int N, int K, int ldc, int act) {
    __shared__ __align__(16) __bf16 BT[128][40];   // [n][k], row stride 80B (16B multiple)

    const int tid  = threadIdx.x;
    const int wave = tid >> 5;          // 0..7
    const int lane = tid & 31;
    const int l16  = lane & 15;
    const int half = lane >> 4;         // 0/1
    const int m0   = blockIdx.y * 64;
    const int n0   = blockIdx.x * 128;
    const int wm   = (wave & 1) * 32;   // 0,32
    const int wn   = (wave >> 1) * 32;  // 0,32,64,96

    f32x8 acc00 = {}, acc01 = {}, acc10 = {}, acc11 = {};

    // clamp rows instead of masking: phantom rows compute garbage that is never stored
    const int ar0 = min(m0 + wm + l16,      M - 1);
    const int ar1 = min(m0 + wm + 16 + l16, M - 1);
    const __bf16* Arow0 = A + (size_t)ar0 * K;
    const __bf16* Arow1 = A + (size_t)ar1 * K;

    // B staging: each thread loads B[kk..kk+1][n0+nn..+7] and stores packed pairs
    const int kk = (tid >> 4) * 2;      // 0,2,..,30
    const int nn = (tid & 15) * 8;      // 0,8,..,120
    const __bf16* bptr = B + (size_t)kk * N + n0 + nn;

    for (int k0 = 0; k0 < K; k0 += 32) {
        bf16x8 brow0 = *(const bf16x8*)bptr;
        bf16x8 brow1 = *(const bf16x8*)(bptr + N);
        bptr += (size_t)32 * N;
        __builtin_prefetch(bptr, 0, 1);                 // next B tile rows
        #pragma unroll
        for (int j = 0; j < 8; ++j) {
            bf16x2 p; p.x = brow0[j]; p.y = brow1[j];
            *(bf16x2*)(&BT[nn + j][kk]) = p;            // ds_store_b32
        }
        __syncthreads();

        // A fragments (lane half selects contiguous K = h*16 .. +15, matching B)
        bf16x8 a0lo = *(const bf16x8*)(Arow0 + k0 + half * 16);
        bf16x8 a0hi = *(const bf16x8*)(Arow0 + k0 + half * 16 + 8);
        bf16x8 a1lo = *(const bf16x8*)(Arow1 + k0 + half * 16);
        bf16x8 a1hi = *(const bf16x8*)(Arow1 + k0 + half * 16 + 8);
        bf16x16 a0 = __builtin_shufflevector(a0lo, a0hi, 0,1,2,3,4,5,6,7,8,9,10,11,12,13,14,15);
        bf16x16 a1 = __builtin_shufflevector(a1lo, a1hi, 0,1,2,3,4,5,6,7,8,9,10,11,12,13,14,15);

        // B fragments: column n, K = half*16 .. +15, contiguous in BT
        bf16x16 b0 = *(const bf16x16*)(&BT[wn + l16][half * 16]);
        bf16x16 b1 = *(const bf16x16*)(&BT[wn + 16 + l16][half * 16]);

        acc00 = __builtin_amdgcn_wmma_f32_16x16x32_bf16(false, a0, false, b0, (short)0, acc00, false, false);
        acc01 = __builtin_amdgcn_wmma_f32_16x16x32_bf16(false, a0, false, b1, (short)0, acc01, false, false);
        acc10 = __builtin_amdgcn_wmma_f32_16x16x32_bf16(false, a1, false, b0, (short)0, acc10, false, false);
        acc11 = __builtin_amdgcn_wmma_f32_16x16x32_bf16(false, a1, false, b1, (short)0, acc11, false, false);
        __syncthreads();
    }

    // epilogue: VGPR v of a 16x16 tile -> row base + half*8 + v, col base + l16
    const int c0 = n0 + wn + l16;
    const int c1 = c0 + 16;
    float bia0 = bias ? bias[c0] : 0.f;
    float bia1 = bias ? bias[c1] : 0.f;
    #pragma unroll
    for (int v = 0; v < 8; ++v) {
        int r0 = m0 + wm + half * 8 + v;
        int r1 = r0 + 16;
        if (r0 < M) {
            float x0 = acc00[v] + bia0, x1 = acc01[v] + bia1;
            if (act) { x0 = fmaxf(x0, 0.f); x1 = fmaxf(x1, 0.f); }
            if (C)  { C[(size_t)r0 * ldc + c0] = x0;  C[(size_t)r0 * ldc + c1] = x1; }
            if (Cb) { Cb[(size_t)r0 * N + c0] = (__bf16)x0; Cb[(size_t)r0 * N + c1] = (__bf16)x1; }
        }
        if (r1 < M) {
            float x0 = acc10[v] + bia0, x1 = acc11[v] + bia1;
            if (act) { x0 = fmaxf(x0, 0.f); x1 = fmaxf(x1, 0.f); }
            if (C)  { C[(size_t)r1 * ldc + c0] = x0;  C[(size_t)r1 * ldc + c1] = x1; }
            if (Cb) { Cb[(size_t)r1 * N + c0] = (__bf16)x0; Cb[(size_t)r1 * N + c1] = (__bf16)x1; }
        }
    }
}

// ---------------------------------------------------------------- edge scatter-add
// 4 edges per block; each thread handles 4 consecutive features of one edge.
__global__ void edge_scatter_add(const int* __restrict__ src, const int* __restrict__ dst,
                                 const float* __restrict__ m, float* __restrict__ agg) {
    int t = threadIdx.x;                  // 256
    int e = blockIdx.x * 4 + (t >> 6);
    int f = (t & 63) * 4;
    int s = src[e], d = dst[e];
    const float4 v = *(const float4*)&m[(size_t)s * HID + f];
    float* o = &agg[(size_t)d * HID + f];
    atomicAdd(o + 0, v.x); atomicAdd(o + 1, v.y);
    atomicAdd(o + 2, v.z); atomicAdd(o + 3, v.w);
}

// ---------------------------------------------------------------- combine + BN stats
#define ROWS_PER_BLK 100
__global__ void gcn_combine_stats(const float* __restrict__ agg, const float* __restrict__ rm,
                                  const float* __restrict__ bias, float* __restrict__ pre,
                                  float* __restrict__ sum, float* __restrict__ sumsq, int rows) {
    int col = threadIdx.x;                 // 256
    int r0 = blockIdx.x * ROWS_PER_BLK;
    int r1 = min(r0 + ROWS_PER_BLK, rows);
    float b = bias[col];
    float s = 0.f, s2 = 0.f;
    for (int r = r0; r < r1; ++r) {
        size_t idx = (size_t)r * HID + col;
        float v = fmaxf(agg[idx] + b, 0.f) + rm[idx];   // rm already relu'd by GEMM epilogue
        pre[idx] = v; s += v; s2 += v * v;
    }
    atomicAdd(&sum[col], s);
    atomicAdd(&sumsq[col], s2);
}

__global__ void bn_apply(const float* __restrict__ pre, const float* __restrict__ sum,
                         const float* __restrict__ sumsq, const float* __restrict__ g,
                         const float* __restrict__ be, float* __restrict__ h,
                         __bf16* __restrict__ hb, int rows) {
    int col = threadIdx.x;
    float inv = 1.f / (float)rows;
    float mu  = sum[col] * inv;
    float var = sumsq[col] * inv - mu * mu;
    float sc  = g[col] * rsqrtf(var + 1e-5f);
    float sh  = be[col] - mu * sc;
    for (int r = blockIdx.x; r < rows; r += gridDim.x) {
        size_t idx = (size_t)r * HID + col;
        float v = pre[idx] * sc + sh;
        if (h)  h[idx] = v;
        hb[idx] = (__bf16)v;
    }
}

// ---------------------------------------------------------------- readout
__device__ inline void atomicMaxFloat(float* addr, float val) {
    unsigned int* ua = (unsigned int*)addr;
    unsigned int old = *ua;
    while (__uint_as_float(old) < val) {
        unsigned int assumed = old;
        old = atomicCAS(ua, assumed, __float_as_uint(val));
        if (old == assumed) break;
    }
}

__global__ void init_z(float* __restrict__ z) {
    int g = blockIdx.x, t = threadIdx.x;            // 512 x 256
    z[(size_t)g * COMB_IN + t] = 0.f;
    z[(size_t)g * COMB_IN + 256 + t] = -FLT_MAX;
}

__global__ void readout(const float* __restrict__ h, const int* __restrict__ gid,
                        const float* __restrict__ awW, const float* __restrict__ awb,
                        float* __restrict__ z) {
    __shared__ float red[8];
    int n = blockIdx.x;
    int t = threadIdx.x;                            // 256
    float hv = h[(size_t)n * HID + t];
    float p = hv * awW[t];
    #pragma unroll
    for (int off = 16; off > 0; off >>= 1) p += __shfl_down(p, off, 32);
    if ((t & 31) == 0) red[t >> 5] = p;
    __syncthreads();
    if (t == 0) {
        float s = 0.f;
        #pragma unroll
        for (int i = 0; i < 8; ++i) s += red[i];
        red[0] = s;
    }
    __syncthreads();
    float w = 1.f / (1.f + __expf(-(red[0] + awb[0])));
    int g = gid[n];
    atomicAdd(&z[(size_t)g * COMB_IN + t], hv * w);
    atomicMaxFloat(&z[(size_t)g * COMB_IN + 256 + t], hv);
}

// ---------------------------------------------------------------- final matvec head
__global__ void final_head(const float* __restrict__ z2, const float* __restrict__ W,
                           const float* __restrict__ b, float* __restrict__ out) {
    __shared__ float red[8];
    int row = blockIdx.x;                           // 512
    int t = threadIdx.x;                            // 256
    float p = z2[(size_t)row * COMB_H2 + t] * W[t]
            + z2[(size_t)row * COMB_H2 + 256 + t] * W[256 + t];
    #pragma unroll
    for (int off = 16; off > 0; off >>= 1) p += __shfl_down(p, off, 32);
    if ((t & 31) == 0) red[t >> 5] = p;
    __syncthreads();
    if (t == 0) {
        float s = 0.f;
        #pragma unroll
        for (int i = 0; i < 8; ++i) s += red[i];
        out[row] = s + b[0];
    }
}

// ---------------------------------------------------------------- host
static inline void launch_gemm(const __bf16* A, const __bf16* B, const float* bias,
                               float* C, __bf16* Cb, int M, int N, int K, int ldc, int act,
                               hipStream_t s) {
    dim3 grid(N / 128, (M + 63) / 64);
    gemm_bf16_wmma<<<grid, 256, 0, s>>>(A, B, bias, C, Cb, M, N, K, ldc, act);
}

extern "C" void kernel_launch(void* const* d_in, const int* in_sizes, int n_in,
                              void* d_out, int out_size, void* d_ws, size_t ws_size,
                              hipStream_t stream) {
    (void)in_sizes; (void)n_in; (void)out_size; (void)ws_size;

    const float* node_feats  = (const float*)d_in[0];
    const float* rdkit_feats = (const float*)d_in[1];
    const int*   src         = (const int*)d_in[2];
    const int*   dst         = (const int*)d_in[3];
    const int*   node_gid    = (const int*)d_in[4];
    const float *W1 = (const float*)d_in[5],  *b1 = (const float*)d_in[6];
    const float *rW1 = (const float*)d_in[7], *rb1 = (const float*)d_in[8];
    const float *g1 = (const float*)d_in[9],  *be1 = (const float*)d_in[10];
    const float *W2 = (const float*)d_in[11], *b2 = (const float*)d_in[12];
    const float *rW2 = (const float*)d_in[13],*rb2 = (const float*)d_in[14];
    const float *g2 = (const float*)d_in[15], *be2 = (const float*)d_in[16];
    const float *aw_W = (const float*)d_in[17], *aw_b = (const float*)d_in[18];
    const float *rk_W1 = (const float*)d_in[19], *rk_b1 = (const float*)d_in[20];
    const float *rk_W2 = (const float*)d_in[21], *rk_b2 = (const float*)d_in[22];
    const float *c_W1 = (const float*)d_in[23], *c_b1 = (const float*)d_in[24];
    const float *c_W2 = (const float*)d_in[25], *c_b2 = (const float*)d_in[26];
    const float *c_W3 = (const float*)d_in[27], *c_b3 = (const float*)d_in[28];
    float* out = (float*)d_out;

    // workspace carve-out
    char* ws = (char*)d_ws;
    size_t off = 0;
    auto alloc = [&](size_t bytes) -> void* {
        void* p = ws + off;
        off += (bytes + 255) & ~(size_t)255;
        return p;
    };
    __bf16* Xb    = (__bf16*)alloc((size_t)N_NODES * IN_FEATS * 2);
    __bf16* W1b   = (__bf16*)alloc((size_t)IN_FEATS * HID * 2);
    __bf16* rW1b  = (__bf16*)alloc((size_t)IN_FEATS * HID * 2);
    __bf16* W2b   = (__bf16*)alloc((size_t)HID * HID * 2);
    __bf16* rW2b  = (__bf16*)alloc((size_t)HID * HID * 2);
    __bf16* rkW1b = (__bf16*)alloc((size_t)RDKIT_IN * RDK_H1 * 2);
    __bf16* rkW2b = (__bf16*)alloc((size_t)RDK_H1 * RDK_H2 * 2);
    __bf16* cW1b  = (__bf16*)alloc((size_t)COMB_IN * COMB_H1 * 2);
    __bf16* cW2b  = (__bf16*)alloc((size_t)COMB_H1 * COMB_H2 * 2);
    float*  bufA  = (float*)alloc((size_t)N_NODES * HID * 4);   // m / pre
    float*  bufB  = (float*)alloc((size_t)N_NODES * HID * 4);   // agg
    float*  bufC  = (float*)alloc((size_t)N_NODES * HID * 4);   // residual (relu'd)
    float*  hF    = (float*)alloc((size_t)N_NODES * HID * 4);   // layer-2 output f32
    __bf16* hb    = (__bf16*)alloc((size_t)N_NODES * HID * 2);
    float*  stats = (float*)alloc(2 * HID * 4);                 // sum | sumsq
    __bf16* rdb   = (__bf16*)alloc((size_t)N_GRAPHS * RDKIT_IN * 2);
    float*  r1    = (float*)alloc((size_t)N_GRAPHS * RDK_H1 * 4);
    __bf16* r1b   = (__bf16*)alloc((size_t)N_GRAPHS * RDK_H1 * 2);
    float*  z     = (float*)alloc((size_t)N_GRAPHS * COMB_IN * 4);
    __bf16* zb    = (__bf16*)alloc((size_t)N_GRAPHS * COMB_IN * 2);
    float*  z1    = (float*)alloc((size_t)N_GRAPHS * COMB_H1 * 4);
    __bf16* z1b   = (__bf16*)alloc((size_t)N_GRAPHS * COMB_H1 * 2);
    float*  z2    = (float*)alloc((size_t)N_GRAPHS * COMB_H2 * 4);

    auto cast = [&](const float* s, __bf16* d, int n) {
        int blocks = (n + 255) / 256; if (blocks > 4096) blocks = 4096;
        cast_f32_bf16<<<blocks, 256, 0, stream>>>(s, d, n);
    };

    // ---- casts (every call; deterministic)
    cast(node_feats, Xb, N_NODES * IN_FEATS);
    cast(rdkit_feats, rdb, N_GRAPHS * RDKIT_IN);
    cast(W1, W1b, IN_FEATS * HID);   cast(rW1, rW1b, IN_FEATS * HID);
    cast(W2, W2b, HID * HID);        cast(rW2, rW2b, HID * HID);
    cast(rk_W1, rkW1b, RDKIT_IN * RDK_H1);
    cast(rk_W2, rkW2b, RDK_H1 * RDK_H2);
    cast(c_W1, cW1b, COMB_IN * COMB_H1);
    cast(c_W2, cW2b, COMB_H1 * COMB_H2);

    const int nStatBlk = (N_NODES + ROWS_PER_BLK - 1) / ROWS_PER_BLK;

    // ---- GCN layer 1
    launch_gemm(Xb, W1b, nullptr, bufA, nullptr, N_NODES, HID, IN_FEATS, HID, 0, stream);
    launch_gemm(Xb, rW1b, rb1,    bufC, nullptr, N_NODES, HID, IN_FEATS, HID, 1, stream);
    hipMemsetAsync(bufB, 0, (size_t)N_NODES * HID * 4, stream);
    hipMemsetAsync(stats, 0, 2 * HID * 4, stream);
    edge_scatter_add<<<N_EDGES / 4, 256, 0, stream>>>(src, dst, bufA, bufB);
    gcn_combine_stats<<<nStatBlk, HID, 0, stream>>>(bufB, bufC, b1, bufA, stats, stats + HID, N_NODES);
    bn_apply<<<256, HID, 0, stream>>>(bufA, stats, stats + HID, g1, be1, nullptr, hb, N_NODES);

    // ---- GCN layer 2
    launch_gemm(hb, W2b, nullptr, bufA, nullptr, N_NODES, HID, HID, HID, 0, stream);
    launch_gemm(hb, rW2b, rb2,    bufC, nullptr, N_NODES, HID, HID, HID, 1, stream);
    hipMemsetAsync(bufB, 0, (size_t)N_NODES * HID * 4, stream);
    hipMemsetAsync(stats, 0, 2 * HID * 4, stream);
    edge_scatter_add<<<N_EDGES / 4, 256, 0, stream>>>(src, dst, bufA, bufB);
    gcn_combine_stats<<<nStatBlk, HID, 0, stream>>>(bufB, bufC, b2, bufA, stats, stats + HID, N_NODES);
    bn_apply<<<256, HID, 0, stream>>>(bufA, stats, stats + HID, g2, be2, hF, hb, N_NODES);

    // ---- readout into z[:, 0:512]
    init_z<<<N_GRAPHS, 256, 0, stream>>>(z);
    readout<<<N_NODES, HID, 0, stream>>>(hF, node_gid, aw_W, aw_b, z);

    // ---- RDKit MLP -> z[:, 512:1024]
    launch_gemm(rdb, rkW1b, rk_b1, r1, r1b, N_GRAPHS, RDK_H1, RDKIT_IN, RDK_H1, 1, stream);
    launch_gemm(r1b, rkW2b, rk_b2, z + 2 * HID, nullptr, N_GRAPHS, RDK_H2, RDK_H1, COMB_IN, 1, stream);

    // ---- combined head
    cast(z, zb, N_GRAPHS * COMB_IN);
    launch_gemm(zb, cW1b, c_b1, z1, z1b, N_GRAPHS, COMB_H1, COMB_IN, COMB_H1, 1, stream);
    launch_gemm(z1b, cW2b, c_b2, z2, nullptr, N_GRAPHS, COMB_H2, COMB_H1, COMB_H2, 1, stream);
    final_head<<<N_GRAPHS, 256, 0, stream>>>(z2, c_W3, c_b3, out);
}
